// LSTM_64854006169635
// MI455X (gfx1250) — matrix operand order repaired and checked by reading, not compiled, not run on hardware
//
#include <hip/hip_runtime.h>
#include <hip/hip_bf16.h>
#include <math.h>

typedef __bf16 bf16;
typedef __attribute__((ext_vector_type(16))) __bf16 v16bf;
typedef __attribute__((ext_vector_type(8)))  __bf16 v8bf;
typedef __attribute__((ext_vector_type(8)))  float  v8f;

#define BDIM 128
#define TDIM 256
#define DDIM 256
#define HDIM 1024
#define CDIM 1000
#define NGATE 4096          // 4*H
#define KTOT 1280           // H + D
#define NPAD 1024           // padded class dim for logits GEMM

#if __has_builtin(__builtin_amdgcn_global_load_async_to_lds_b128)
#define USE_ASYNC_LDS 1
#else
#define USE_ASYNC_LDS 0
#endif

// async-to-LDS builtin parameter types: int4 in AS(1) / AS(3)
typedef int v4i_ __attribute__((vector_size(16)));
typedef __attribute__((address_space(1))) v4i_ gv4i;
typedef __attribute__((address_space(3))) v4i_ lv4i;

__device__ __forceinline__ void wait_async_zero() {
#if __has_builtin(__builtin_amdgcn_s_wait_asynccnt)
    __builtin_amdgcn_s_wait_asynccnt(0);
#else
    asm volatile("s_wait_asynccnt 0x0" ::: "memory");
#endif
}

// ---------------- GEMM: C[M x N] = A[M x K] * Bt[N x K]^T  (bf16 in, f32 out) ----
// A split along K into A0 (K < K0, stride lda0) and A1 (K >= K0, stride lda1).
// Bt is pre-transposed (N-major, stride ldb).
// Block tile 64x128, 8 waves in 2(M) x 4(N); each wave computes 32x32
// (4 WMMA accumulators). Double-buffered LDS, one barrier per K-chunk.
// Tile fill uses GLOBAL_LOAD_ASYNC_TO_LDS_B128 (ASYNCcnt) when available.
#define BM 64
#define BN 128
#define BK 32
#define LDSS 40   // padded bf16 row stride in LDS (80 B, 16B-aligned)

__global__ __launch_bounds__(256) void wmma_gemm_kernel(
    const bf16* __restrict__ A0, int lda0, int K0,
    const bf16* __restrict__ A1, int lda1,
    const bf16* __restrict__ Bt, int ldb,
    float* __restrict__ Cout, int ldc,
    int Ktot)
{
    __shared__ bf16 sA[2 * BM * LDSS];   // 10 KB
    __shared__ bf16 sB[2 * BN * LDSS];   // 20 KB

    const int tid  = threadIdx.x;
    const int lane = tid & 31;
    const int wave = tid >> 5;
    const int wm   = (wave & 1) * 32;    // 0 or 32
    const int wn   = (wave >> 1) * 32;   // 0,32,64,96

    const int bm = blockIdx.y * BM;
    const int bn = blockIdx.x * BN;

    const int half = lane >> 4;          // 0 or 1
    const int l16  = lane & 15;

    // cooperative fill coords: A 64x32 (8 bf16/thread), B 128x32 (16 bf16/thread)
    const int arow = tid >> 2;           // 0..63
    const int acol = (tid & 3) * 8;      // 0,8,16,24
    const int brow = tid >> 1;           // 0..127
    const int bcol = (tid & 1) * 16;     // 0,16

    // lane-invariant LDS fragment offsets (in bf16 elements)
    const int aoff0 = (wm + l16) * LDSS;             // A tile row base
    const int aoff1 = aoff0 + 16 * LDSS;
    const int boff0 = (wn + l16) * LDSS;             // B tile row base (N-major)
    const int boff1 = boff0 + 16 * LDSS;
    const int aeo0  = half * 8;                      // A elems 0..7  : K = half*8+e
    const int aeo1  = 16 + half * 8;                 // A elems 8..15 : K = 16+half*8+e
    const int beo   = half * 8;                      // B elems 0..7 : K = half*8+e; 8..15 : K = 16+half*8+e

    const int nch = Ktot / BK;

    v8f acc00 = {}, acc01 = {}, acc10 = {}, acc11 = {};

#if USE_ASYNC_LDS
    // issue per-lane async DMA global->LDS for chunk kb into buffer `buf`
    auto fill = [&](int kb, int buf) {
        const bf16* aptr = (kb < K0)
            ? A0 + (size_t)(bm + arow) * lda0 + (kb + acol)
            : A1 + (size_t)(bm + arow) * lda1 + (kb - K0 + acol);
        bf16* adst = &sA[buf * (BM * LDSS) + arow * LDSS + acol];
        __builtin_amdgcn_global_load_async_to_lds_b128(
            (gv4i*)aptr, (lv4i*)adst, 0, 0);
        const bf16* bptr = Bt + (size_t)(bn + brow) * ldb + (kb + bcol);
        bf16* bdst = &sB[buf * (BN * LDSS) + brow * LDSS + bcol];
        __builtin_amdgcn_global_load_async_to_lds_b128(
            (gv4i*)bptr, (lv4i*)bdst, 0, 0);
        __builtin_amdgcn_global_load_async_to_lds_b128(
            (gv4i*)(bptr + 8), (lv4i*)(bdst + 8), 0, 0);
    };
    auto settle = [&]() { wait_async_zero(); };
#else
    v8bf ra, rb0, rb1;                   // global->LDS staging registers
    auto fill = [&](int kb, int buf) {
        const bf16* aptr = (kb < K0)
            ? A0 + (size_t)(bm + arow) * lda0 + (kb + acol)
            : A1 + (size_t)(bm + arow) * lda1 + (kb - K0 + acol);
        ra = *(const v8bf*)aptr;
        const bf16* bptr = Bt + (size_t)(bn + brow) * ldb + (kb + bcol);
        rb0 = *(const v8bf*)bptr;
        rb1 = *(const v8bf*)(bptr + 8);
        *(v8bf*)&sA[buf * (BM * LDSS) + arow * LDSS + acol] = ra;
        bf16* d = &sB[buf * (BN * LDSS) + brow * LDSS + bcol];
        *(v8bf*)d       = rb0;
        *(v8bf*)(d + 8) = rb1;
    };
    auto settle = [&]() {};
#endif

    // prologue: chunk 0 -> buffer 0
    fill(0, 0);
    settle();
    __syncthreads();

    for (int i = 0; i < nch; ++i) {
        const int cur = i & 1;
        if (i + 1 < nch) {
            fill((i + 1) * BK, cur ^ 1);          // DMA next chunk under compute
            if (i + 2 < nch) {                    // pull B stream 2 chunks ahead (L2)
                __builtin_prefetch(Bt + (size_t)(bn + brow) * ldb +
                                   ((i + 2) * BK + bcol), 0, 1);
            }
        }

        const bf16* cA = &sA[cur * (BM * LDSS)];
        const bf16* cB = &sB[cur * (BN * LDSS)];

        v16bf a0, a1, b0, b1;
        ((v8bf*)&a0)[0] = *(const v8bf*)(cA + aoff0 + aeo0);
        ((v8bf*)&a0)[1] = *(const v8bf*)(cA + aoff0 + aeo1);
        ((v8bf*)&a1)[0] = *(const v8bf*)(cA + aoff1 + aeo0);
        ((v8bf*)&a1)[1] = *(const v8bf*)(cA + aoff1 + aeo1);
        ((v8bf*)&b0)[0] = *(const v8bf*)(cB + boff0 + beo);
        ((v8bf*)&b0)[1] = *(const v8bf*)(cB + boff0 + beo + 16);
        ((v8bf*)&b1)[0] = *(const v8bf*)(cB + boff1 + beo);
        ((v8bf*)&b1)[1] = *(const v8bf*)(cB + boff1 + beo + 16);

        acc00 = __builtin_amdgcn_wmma_f32_16x16x32_bf16(false, a0, false, b0,
                                                        (short)0, acc00, false, false);
        acc01 = __builtin_amdgcn_wmma_f32_16x16x32_bf16(false, a0, false, b1,
                                                        (short)0, acc01, false, false);
        acc10 = __builtin_amdgcn_wmma_f32_16x16x32_bf16(false, a1, false, b0,
                                                        (short)0, acc10, false, false);
        acc11 = __builtin_amdgcn_wmma_f32_16x16x32_bf16(false, a1, false, b1,
                                                        (short)0, acc11, false, false);

        if (i + 1 < nch) {
            settle();                    // ASYNCcnt==0: next buffer fully in LDS
            __syncthreads();             // single barrier per K-chunk
        }
    }

    // D layout: VGPR r -> M = r (lanes 0-15) or 8+r (lanes 16-31); N = lane&15
    const int mb0 = bm + wm + (half ? 8 : 0);
    const int n0  = bn + wn + l16;
#pragma unroll
    for (int r = 0; r < 8; ++r) {
        Cout[(size_t)(mb0 + r) * ldc + n0]           = acc00[r];
        Cout[(size_t)(mb0 + r) * ldc + n0 + 16]      = acc01[r];
        Cout[(size_t)(mb0 + 16 + r) * ldc + n0]      = acc10[r];
        Cout[(size_t)(mb0 + 16 + r) * ldc + n0 + 16] = acc11[r];
    }
}

// ---------------- prep kernels ----------------------------------------------
__global__ __launch_bounds__(256) void cvt_x_kernel(const float* __restrict__ x,
                                                    bf16* __restrict__ xb)
{
    int i = (blockIdx.x * 256 + threadIdx.x) * 4;
#pragma unroll
    for (int k = 0; k < 4; ++k) xb[i + k] = (bf16)x[i + k];
}

// Whx[n][k], n = gate*H + j (gate order g,i,f,o), k<H from W_?h[k][j], else W_?x[k-H][j]
__global__ __launch_bounds__(256) void build_whx_kernel(
    const float* __restrict__ Wgh, const float* __restrict__ Wih,
    const float* __restrict__ Wfh, const float* __restrict__ Woh,
    const float* __restrict__ Wgx, const float* __restrict__ Wix,
    const float* __restrict__ Wfx, const float* __restrict__ Wox,
    bf16* __restrict__ Whx)
{
    int idx = blockIdx.x * 256 + threadIdx.x;        // over 4096*1280
    int n = idx / KTOT;
    int k = idx - n * KTOT;
    int gate = n >> 10;
    int j = n & (HDIM - 1);
    const float* Wh = (gate == 0) ? Wgh : (gate == 1) ? Wih : (gate == 2) ? Wfh : Woh;
    const float* Wx = (gate == 0) ? Wgx : (gate == 1) ? Wix : (gate == 2) ? Wfx : Wox;
    float v = (k < HDIM) ? Wh[(size_t)k * HDIM + j]
                         : Wx[(size_t)(k - HDIM) * HDIM + j];
    Whx[idx] = (bf16)v;
}

// Wp_t[n][k] (NPAD x H), n<1000 from W_ph[k][n], else 0
__global__ __launch_bounds__(256) void build_wp_kernel(const float* __restrict__ Wph,
                                                       bf16* __restrict__ Wpt)
{
    int idx = blockIdx.x * 256 + threadIdx.x;        // over 1024*1024
    int n = idx >> 10;
    int k = idx & (HDIM - 1);
    Wpt[idx] = (n < CDIM) ? (bf16)Wph[(size_t)k * CDIM + n] : (bf16)0.0f;
}

__global__ __launch_bounds__(256) void init_state_kernel(float* __restrict__ c,
                                                         bf16* __restrict__ h)
{
    int idx = blockIdx.x * 256 + threadIdx.x;        // over 128*1024
    c[idx] = 0.0f;
    h[idx] = (bf16)0.0f;
}

// ---------------- pointwise LSTM cell update --------------------------------
__global__ __launch_bounds__(256) void lstm_pointwise_kernel(
    const float* __restrict__ Z,          // [128][4096] gate pre-acts (no bias)
    const float* __restrict__ bg, const float* __restrict__ bi,
    const float* __restrict__ bfv, const float* __restrict__ bo,
    float* __restrict__ c, bf16* __restrict__ h)
{
    int idx = blockIdx.x * 256 + threadIdx.x;        // over 128*1024
    int b = idx >> 10;
    int j = idx & (HDIM - 1);
    const float* zr = Z + (size_t)b * NGATE;
    float zg = zr[j]            + bg[j];
    float zi = zr[HDIM + j]     + bi[j];
    float zf = zr[2 * HDIM + j] + bfv[j];
    float zo = zr[3 * HDIM + j] + bo[j];
    float g = tanhf(zg);
    float i = 1.0f / (1.0f + __expf(-zi));
    float f = 1.0f / (1.0f + __expf(-zf));
    float o = 1.0f / (1.0f + __expf(-zo));
    float cn = g * i + c[idx] * f;
    c[idx] = cn;
    h[idx] = (bf16)(tanhf(cn) * o);
}

// ---------------- log-softmax over classes ----------------------------------
__global__ __launch_bounds__(256) void logsoftmax_kernel(
    const float* __restrict__ logits,     // [128][1024] (first 1000 valid)
    const float* __restrict__ bp,
    float* __restrict__ out)              // [128][1000]
{
    __shared__ float red[256];
    const int b = blockIdx.x;
    const float* row = logits + (size_t)b * NPAD;
    float mx = -INFINITY;
    for (int j = threadIdx.x; j < CDIM; j += 256)
        mx = fmaxf(mx, row[j] + bp[j]);
    red[threadIdx.x] = mx;
    __syncthreads();
    for (int s = 128; s > 0; s >>= 1) {
        if (threadIdx.x < s)
            red[threadIdx.x] = fmaxf(red[threadIdx.x], red[threadIdx.x + s]);
        __syncthreads();
    }
    mx = red[0];
    __syncthreads();
    float sum = 0.0f;
    for (int j = threadIdx.x; j < CDIM; j += 256)
        sum += __expf(row[j] + bp[j] - mx);
    red[threadIdx.x] = sum;
    __syncthreads();
    for (int s = 128; s > 0; s >>= 1) {
        if (threadIdx.x < s) red[threadIdx.x] += red[threadIdx.x + s];
        __syncthreads();
    }
    float lse = __logf(red[0]) + mx;
    for (int j = threadIdx.x; j < CDIM; j += 256)
        out[(size_t)b * CDIM + j] = row[j] + bp[j] - lse;
}

// ---------------- host-side orchestration -----------------------------------
extern "C" void kernel_launch(void* const* d_in, const int* in_sizes, int n_in,
                              void* d_out, int out_size, void* d_ws, size_t ws_size,
                              hipStream_t stream)
{
    const float* x    = (const float*)d_in[0];
    const float* Wgx  = (const float*)d_in[1];
    const float* Wix  = (const float*)d_in[2];
    const float* Wfx  = (const float*)d_in[3];
    const float* Wox  = (const float*)d_in[4];
    const float* Wgh  = (const float*)d_in[5];
    const float* Wih  = (const float*)d_in[6];
    const float* Wfh  = (const float*)d_in[7];
    const float* Woh  = (const float*)d_in[8];
    const float* Wph  = (const float*)d_in[9];
    const float* bg   = (const float*)d_in[10];
    const float* bi   = (const float*)d_in[11];
    const float* bfv  = (const float*)d_in[12];
    const float* bo   = (const float*)d_in[13];
    const float* bp   = (const float*)d_in[14];
    float* out = (float*)d_out;

    // workspace carve-up
    char* ws = (char*)d_ws;
    size_t off = 0;
    bf16*  Whx    = (bf16*)(ws + off); off += (size_t)NGATE * KTOT * sizeof(bf16);   // 10 MB
    bf16*  Wpt    = (bf16*)(ws + off); off += (size_t)NPAD * HDIM * sizeof(bf16);    // 2 MB
    bf16*  xb     = (bf16*)(ws + off); off += (size_t)BDIM * TDIM * DDIM * sizeof(bf16); // 16 MB
    bf16*  h      = (bf16*)(ws + off); off += (size_t)BDIM * HDIM * sizeof(bf16);    // 256 KB
    float* c      = (float*)(ws + off); off += (size_t)BDIM * HDIM * sizeof(float);  // 512 KB
    float* Z      = (float*)(ws + off); off += (size_t)BDIM * NGATE * sizeof(float); // 2 MB
    float* logits = (float*)(ws + off); off += (size_t)BDIM * NPAD * sizeof(float);  // 512 KB
    (void)ws_size; (void)in_sizes; (void)n_in; (void)out_size;

    // ---- prep ----
    cvt_x_kernel<<<(BDIM * TDIM * DDIM) / (256 * 4), 256, 0, stream>>>(x, xb);
    build_whx_kernel<<<((size_t)NGATE * KTOT) / 256, 256, 0, stream>>>(
        Wgh, Wih, Wfh, Woh, Wgx, Wix, Wfx, Wox, Whx);
    build_wp_kernel<<<((size_t)NPAD * HDIM) / 256, 256, 0, stream>>>(Wph, Wpt);
    init_state_kernel<<<(BDIM * HDIM) / 256, 256, 0, stream>>>(c, h);

    // ---- recurrent loop: Z = [h | x_t] @ Whx^T, then pointwise cell update ----
    dim3 ggrid(NGATE / BN, BDIM / BM);   // 32 x 2
    for (int t = 0; t < TDIM; ++t) {
        const bf16* xslice = xb + (size_t)t * DDIM;   // row m -> + m*(T*D)
        wmma_gemm_kernel<<<ggrid, 256, 0, stream>>>(
            h, HDIM, HDIM,                 // A0 = h[128][1024]
            xslice, TDIM * DDIM,           // A1 = x_t[128][256], row stride T*D
            Whx, KTOT,                     // Bt = [4096][1280]
            Z, NGATE, KTOT);
        lstm_pointwise_kernel<<<(BDIM * HDIM) / 256, 256, 0, stream>>>(
            Z, bg, bi, bfv, bo, c, h);
    }

    // ---- classifier head ----
    dim3 pgrid(NPAD / BN, BDIM / BM);    // 8 x 2
    wmma_gemm_kernel<<<pgrid, 256, 0, stream>>>(
        h, HDIM, HDIM,
        h, HDIM,                          // unused second region (Ktot == K0)
        Wpt, HDIM,
        logits, NPAD, HDIM);
    logsoftmax_kernel<<<BDIM, 256, 0, stream>>>(logits, bp, out);
}
